// ECDS_69956427317652
// MI455X (gfx1250) — compile-verified, matrix operand order checked by reading, not run on hardware
//
#include <hip/hip_runtime.h>
#include <hip/hip_bf16.h>
#include <stdint.h>

// ============================================================================
// FlowNet3D-style forward pass for MI455X (gfx1250, wave32, WMMA).
// MLP layers: f32 params/activations are pre-converted once per GEMM into
// padded f16 buffers (W row-major, X transposed so K is contiguous). The WMMA
// kernel register-blocks 4 co-tiles per wave, reusing each B (activation)
// fragment across 4 v_wmma_f32_16x16x32_f16 issues:
//   per K-step: 2x b128 (B) + 8x b128 (A) -> 4x v_wmma.
// Distance/top-k/FPS kernels keep data in LDS (<42KB of the 320KB WGP LDS).
// NOTE: workspace layout assumes ws_size >= ~580MB (su2 stage peak).
// ============================================================================

typedef __attribute__((ext_vector_type(16))) _Float16 v16h;
typedef __attribute__((ext_vector_type(8)))  _Float16 v8h;
typedef __attribute__((ext_vector_type(8)))  float    v8f;

#define BN_EPS    1e-5f
#define NEG_SLOPE 0.2f

// ---------------------------------------------------------------------------
// Pre-pass: W (Co x Ci, f32) -> Wp (CoPad x CiPad, f16, zero-padded)
// ---------------------------------------------------------------------------
__global__ void k_cvt_w(const float* __restrict__ W, _Float16* __restrict__ Wp,
                        int Co, int Ci, int CoPad, int CiPad)
{
    int i = blockIdx.x * blockDim.x + threadIdx.x;
    if (i >= CoPad * CiPad) return;
    int r = i / CiPad, c = i - r * CiPad;
    Wp[i] = (r < Co && c < Ci) ? (_Float16)W[(size_t)r * Ci + c] : (_Float16)0.f;
}

// ---------------------------------------------------------------------------
// Pre-pass: X (Ci x M, f32) -> Xt (Mpad x CiPad, f16, zero-padded, transposed)
// Writes are fully coalesced (k fastest); reads strided once.
// ---------------------------------------------------------------------------
__global__ void k_cvt_xt(const float* __restrict__ X, _Float16* __restrict__ Xt,
                         int Ci, int M, int CiPad, int Mpad)
{
    size_t i = (size_t)blockIdx.x * blockDim.x + threadIdx.x;
    size_t total = (size_t)Mpad * CiPad;
    if (i >= total) return;
    int k = (int)(i % CiPad);
    int m = (int)(i / CiPad);
    Xt[i] = (k < Ci && m < M) ? (_Float16)X[(size_t)k * M + m] : (_Float16)0.f;
}

// ---------------------------------------------------------------------------
// WMMA GEMM on packed f16:  Y[co][m] = sum_ci W[co][ci]*X[ci][m] + bias[co]
// Wp: (CoPad x CiPad) f16 row-major;  Xt: (Mpad x CiPad) f16 row-major.
// Block = 128 threads (4 waves). Each wave owns one m-tile (blockIdx.x*4+wave)
// and FOUR co-tiles (blockIdx.y*4 .. +3), reusing the B fragment 4x.
// 16-bit A/B VGPR layout (ISA 7.12.2): lanes 0-15 hold K={0..7,16..23},
// lanes 16-31 hold K={8..15,24..31}; both runs are contiguous 16B in memory.
// ---------------------------------------------------------------------------
__global__ void k_wmma_gemm_f16(const _Float16* __restrict__ Wp,
                                const _Float16* __restrict__ Xt,
                                const float* __restrict__ bias,
                                float* __restrict__ Y,
                                int Co, int CiPad, int M)
{
    const int wave    = threadIdx.x >> 5;
    const int lane    = threadIdx.x & 31;
    const int mTile   = blockIdx.x * 4 + wave;
    const int coTile0 = blockIdx.y * 4;

    const int row = lane & 15;
    const int khi = lane >> 4;

    const _Float16* ap = Wp + (size_t)(coTile0 * 16 + row) * CiPad + khi * 8;
    const _Float16* bp = Xt + (size_t)(mTile * 16 + row) * CiPad + khi * 8;
    const size_t aStride = (size_t)16 * CiPad;   // one co-tile of A rows

    v8f acc[4] = {};

    union Pack { v16h v; v8h h[2]; };

    for (int k0 = 0; k0 < CiPad; k0 += 32) {
        __builtin_prefetch(bp + 32, 0, 1);       // global_prefetch next B slab
        Pack b;
        b.h[0] = *(const v8h*)(bp);              // K = k0 + khi*8 + {0..7}
        b.h[1] = *(const v8h*)(bp + 16);         // K = k0 + 16 + khi*8 + {0..7}
        bp += 32;

        Pack a[4];
#pragma unroll
        for (int t = 0; t < 4; ++t) {
            a[t].h[0] = *(const v8h*)(ap + t * aStride);
            a[t].h[1] = *(const v8h*)(ap + t * aStride + 16);
        }
        ap += 32;

#pragma unroll
        for (int t = 0; t < 4; ++t)
            acc[t] = __builtin_amdgcn_wmma_f32_16x16x32_f16(
                         false, a[t].v, false, b.v, (short)0, acc[t], false, false);
    }

    // C/D layout: lane<16 -> M=v, N=lane ; lane>=16 -> M=v+8, N=lane-16
    const int n = mTile * 16 + row;
#pragma unroll
    for (int t = 0; t < 4; ++t) {
#pragma unroll
        for (int v = 0; v < 8; ++v) {
            int m = (coTile0 + t) * 16 + v + 8 * khi;
            if (m < Co && n < M)
                Y[(size_t)m * (size_t)M + n] = acc[t][v] + bias[m];
        }
    }
}

// ---------------------------------------------------------------------------
// BatchNorm statistics: one block per channel; population mean/var over M.
// ---------------------------------------------------------------------------
__global__ void k_bn_stats(const float* __restrict__ Y, int M, float* __restrict__ stats)
{
    const int c = blockIdx.x;
    const float* row = Y + (size_t)c * M;
    float s1 = 0.f, s2 = 0.f;
    for (int j = threadIdx.x; j < M; j += blockDim.x) {
        float v = row[j]; s1 += v; s2 += v * v;
    }
    __shared__ float sa[256], sb[256];
    sa[threadIdx.x] = s1; sb[threadIdx.x] = s2;
    __syncthreads();
    for (int s = 128; s > 0; s >>= 1) {
        if (threadIdx.x < s) { sa[threadIdx.x] += sa[threadIdx.x + s];
                               sb[threadIdx.x] += sb[threadIdx.x + s]; }
        __syncthreads();
    }
    if (threadIdx.x == 0) {
        float m = sa[0] / (float)M;
        stats[2 * c]     = m;
        stats[2 * c + 1] = sb[0] / (float)M - m * m;
    }
}

// act: 0 = none, 1 = relu
__global__ void k_bn_apply(float* __restrict__ Y, int M,
                           const float* __restrict__ stats,
                           const float* __restrict__ g,
                           const float* __restrict__ beta, int act)
{
    const int c = blockIdx.y;
    const int j = blockIdx.x * blockDim.x + threadIdx.x;
    if (j >= M) return;
    float m = stats[2 * c], v = stats[2 * c + 1];
    float val = (Y[(size_t)c * M + j] - m) * rsqrtf(v + BN_EPS) * g[c] + beta[c];
    if (act == 1) val = fmaxf(val, 0.0f);
    Y[(size_t)c * M + j] = val;
}

// ---------------------------------------------------------------------------
// Farthest point sampling. Single block, iterative. xyz: (C x N), C<=256.
// ---------------------------------------------------------------------------
__global__ void k_fps(const float* __restrict__ xyz, int C, int N,
                      int npoint, int* __restrict__ out_idx)
{
    __shared__ float sdist[8192];
    __shared__ float scen[256];
    __shared__ float rv[1024];
    __shared__ int   ri[1024];
    __shared__ int   sfar;

    const int tid = threadIdx.x, bd = blockDim.x;
    for (int j = tid; j < N; j += bd) sdist[j] = 1e10f;
    if (tid == 0) sfar = 0;
    __syncthreads();

    for (int it = 0; it < npoint; ++it) {
        if (tid == 0) out_idx[it] = sfar;
        for (int c = tid; c < C; c += bd) scen[c] = xyz[(size_t)c * N + sfar];
        __syncthreads();

        float bv = -1.0f; int bi = 0x7fffffff;
        for (int j = tid; j < N; j += bd) {
            float d = 0.f;
            for (int c = 0; c < C; ++c) {
                float t = xyz[(size_t)c * N + j] - scen[c];
                d += t * t;
            }
            float nd = fminf(sdist[j], d);
            sdist[j] = nd;
            if (nd > bv) { bv = nd; bi = j; }
        }
        rv[tid] = bv; ri[tid] = bi;
        __syncthreads();
        for (int s = bd >> 1; s > 0; s >>= 1) {
            if (tid < s) {
                if (rv[tid + s] > rv[tid] ||
                    (rv[tid + s] == rv[tid] && ri[tid + s] < ri[tid])) {
                    rv[tid] = rv[tid + s]; ri[tid] = ri[tid + s];
                }
            }
            __syncthreads();
        }
        if (tid == 0) sfar = ri[0];
        __syncthreads();
    }
}

// ---------------------------------------------------------------------------
// Ball query: one block per query; first K ascending in-radius indices,
// padded with the first hit (matches the reference's score top-k).
// ---------------------------------------------------------------------------
__global__ void k_ball(const float* __restrict__ xyz, int C, int N,
                       const float* __restrict__ newxyz, int Nq,
                       float r2, int K, int* __restrict__ out)
{
    __shared__ float sq[256];
    __shared__ float sd[8192];
    const int q = blockIdx.x, tid = threadIdx.x, bd = blockDim.x;

    for (int c = tid; c < C; c += bd) sq[c] = newxyz[(size_t)c * Nq + q];
    __syncthreads();
    for (int j = tid; j < N; j += bd) {
        float d = 0.f;
        for (int c = 0; c < C; ++c) { float t = sq[c] - xyz[(size_t)c * N + j]; d += t * t; }
        sd[j] = d;
    }
    __syncthreads();
    if (tid == 0) {
        int cnt = 0, first = 0; bool has = false;
        for (int j = 0; j < N && cnt < K; ++j) {
            if (sd[j] <= r2) {
                if (!has) { first = j; has = true; }
                out[(size_t)q * K + cnt++] = j;
            }
        }
        int pad = has ? first : 0;
        for (; cnt < K; ++cnt) out[(size_t)q * K + cnt] = pad;
    }
}

// ---------------------------------------------------------------------------
// kNN: one block per query; distance vector in LDS, K argmin passes
// (smaller index wins ties -> matches jax.lax.top_k stability).
// ---------------------------------------------------------------------------
__global__ void k_knn(const float* __restrict__ query, const float* __restrict__ ref,
                      int C, int Nq, int Nr, int K, int* __restrict__ out)
{
    __shared__ float sq[256];
    __shared__ float sd[8192];
    __shared__ float rv[256];
    __shared__ int   ri[256];
    const int q = blockIdx.x, tid = threadIdx.x, bd = blockDim.x;

    for (int c = tid; c < C; c += bd) sq[c] = query[(size_t)c * Nq + q];
    __syncthreads();
    for (int j = tid; j < Nr; j += bd) {
        float d = 0.f;
        for (int c = 0; c < C; ++c) { float t = sq[c] - ref[(size_t)c * Nr + j]; d += t * t; }
        sd[j] = d;
    }
    __syncthreads();
    for (int t = 0; t < K; ++t) {
        float bv = 3.4e38f; int bi = 0x7fffffff;
        for (int j = tid; j < Nr; j += bd) {
            float d = sd[j];
            if (d < bv || (d == bv && j < bi)) { bv = d; bi = j; }
        }
        rv[tid] = bv; ri[tid] = bi;
        __syncthreads();
        for (int s = bd >> 1; s > 0; s >>= 1) {
            if (tid < s) {
                if (rv[tid + s] < rv[tid] ||
                    (rv[tid + s] == rv[tid] && ri[tid + s] < ri[tid])) {
                    rv[tid] = rv[tid + s]; ri[tid] = ri[tid + s];
                }
            }
            __syncthreads();
        }
        if (tid == 0) { out[(size_t)q * K + t] = ri[0]; sd[ri[0]] = 3.4e38f; }
        __syncthreads();
    }
}

// ---------------------------------------------------------------------------
// Grouped gather: dst[c][q][k] = src[c][idx[q*K+k]] (- center[c][q] if sub)
// ---------------------------------------------------------------------------
__global__ void k_group_gather(const float* __restrict__ src, int C, int N,
                               const int* __restrict__ idx, int Nq, int K,
                               float* __restrict__ dst, int sub,
                               const float* __restrict__ center)
{
    size_t t = (size_t)blockIdx.x * blockDim.x + threadIdx.x;
    size_t total = (size_t)C * Nq * K;
    if (t >= total) return;
    int k = (int)(t % K);
    size_t iq = t / K;
    int q = (int)(iq % Nq);
    int c = (int)(iq / Nq);
    float v = src[(size_t)c * N + idx[(size_t)q * K + k]];
    if (sub) v -= center[(size_t)c * Nq + q];
    dst[t] = v;
}

__global__ void k_gather_cols(const float* __restrict__ src, int C, int N,
                              const int* __restrict__ idx, int Nq,
                              float* __restrict__ dst)
{
    int i = blockIdx.x * blockDim.x + threadIdx.x;
    if (i >= C * Nq) return;
    int c = i / Nq, q = i - c * Nq;
    dst[(size_t)c * Nq + q] = src[(size_t)c * N + idx[q]];
}

__global__ void k_broadcastK(const float* __restrict__ src, int C, int Nq, int K,
                             float* __restrict__ dst)
{
    size_t t = (size_t)blockIdx.x * blockDim.x + threadIdx.x;
    size_t total = (size_t)C * Nq * K;
    if (t >= total) return;
    size_t iq = t / K;
    int q = (int)(iq % Nq);
    int c = (int)(iq / Nq);
    dst[t] = src[(size_t)c * Nq + q];
}

__global__ void k_maxK(const float* __restrict__ src, int C, int Nq, int K,
                       float* __restrict__ dst)
{
    int i = blockIdx.x * blockDim.x + threadIdx.x;
    if (i >= C * Nq) return;
    const float* p = src + (size_t)i * K;
    float m = p[0];
    for (int k = 1; k < K; ++k) m = fmaxf(m, p[k]);
    dst[i] = m;
}

// ---------------------------------------------------------------------------
// feature_prop 3-NN inverse-distance interpolation. One block per query point.
// ---------------------------------------------------------------------------
__global__ void k_interp3(const float* __restrict__ pos1, const float* __restrict__ pos2,
                          const float* __restrict__ feat2, int N1, int N2, int C2,
                          float* __restrict__ out)
{
    __shared__ float sd[8192];
    __shared__ float rv[256];
    __shared__ int   ri[256];
    __shared__ int   sidx[3];
    __shared__ float sw[3];
    const int q = blockIdx.x, tid = threadIdx.x, bd = blockDim.x;

    float qx = pos1[q], qy = pos1[(size_t)N1 + q], qz = pos1[2 * (size_t)N1 + q];
    for (int j = tid; j < N2; j += bd) {
        float dx = qx - pos2[j];
        float dy = qy - pos2[(size_t)N2 + j];
        float dz = qz - pos2[2 * (size_t)N2 + j];
        sd[j] = dx * dx + dy * dy + dz * dz;
    }
    __syncthreads();
    for (int t = 0; t < 3; ++t) {
        float bv = 3.4e38f; int bi = 0x7fffffff;
        for (int j = tid; j < N2; j += bd) {
            float d = sd[j];
            if (d < bv || (d == bv && j < bi)) { bv = d; bi = j; }
        }
        rv[tid] = bv; ri[tid] = bi;
        __syncthreads();
        for (int s = bd >> 1; s > 0; s >>= 1) {
            if (tid < s) {
                if (rv[tid + s] < rv[tid] ||
                    (rv[tid + s] == rv[tid] && ri[tid + s] < ri[tid])) {
                    rv[tid] = rv[tid + s]; ri[tid] = ri[tid + s];
                }
            }
            __syncthreads();
        }
        if (tid == 0) { sidx[t] = ri[0]; sw[t] = rv[0]; sd[ri[0]] = 3.4e38f; }
        __syncthreads();
    }
    if (tid == 0) {
        float w0 = 1.f / (fmaxf(sw[0], 0.f) + 1e-8f);
        float w1 = 1.f / (fmaxf(sw[1], 0.f) + 1e-8f);
        float w2 = 1.f / (fmaxf(sw[2], 0.f) + 1e-8f);
        float s = w0 + w1 + w2;
        sw[0] = w0 / s; sw[1] = w1 / s; sw[2] = w2 / s;
    }
    __syncthreads();
    int i0 = sidx[0], i1 = sidx[1], i2 = sidx[2];
    float w0 = sw[0], w1 = sw[1], w2 = sw[2];
    for (int c = tid; c < C2; c += bd) {
        out[(size_t)c * N1 + q] =
            feat2[(size_t)c * N2 + i0] * w0 +
            feat2[(size_t)c * N2 + i1] * w1 +
            feat2[(size_t)c * N2 + i2] * w2;
    }
}

__global__ void k_leaky(float* __restrict__ y, int n)
{
    int i = blockIdx.x * blockDim.x + threadIdx.x;
    if (i >= n) return;
    float v = y[i];
    y[i] = (v > 0.f) ? v : NEG_SLOPE * v;
}

__global__ void k_add(const float* __restrict__ a, const float* __restrict__ b,
                      float* __restrict__ o, int n)
{
    int i = blockIdx.x * blockDim.x + threadIdx.x;
    if (i >= n) return;
    o[i] = a[i] + b[i];
}

// ============================================================================
// Host orchestration
// ============================================================================
struct Layer { const float *W, *b, *g, *beta; int ci, co; };

static inline int ceilN(int x, int n) { return ((x + n - 1) / n) * n; }

extern "C" void kernel_launch(void* const* d_in, const int* in_sizes, int n_in,
                              void* d_out, int out_size, void* d_ws, size_t ws_size,
                              hipStream_t stream)
{
    (void)in_sizes; (void)n_in; (void)out_size; (void)ws_size;

    const float* x = (const float*)d_in[0];       // (1,4,3,8192)
    const float* x0 = x;
    const float* x1 = x + 1 * 3 * 8192;
    const float* x2 = x + 2 * 3 * 8192;
    const float* x3 = x + 3 * 3 * 8192;

    // --- unpack params (setup_inputs insertion order; per layer: W,b,g,beta)
    int pi = 1;
    auto take = [&](Layer* Ls, int n, const int* dims) {
        for (int l = 0; l < n; ++l) {
            Ls[l].ci = dims[l]; Ls[l].co = dims[l + 1];
            Ls[l].W    = (const float*)d_in[pi++];
            Ls[l].b    = (const float*)d_in[pi++];
            Ls[l].g    = (const float*)d_in[pi++];
            Ls[l].beta = (const float*)d_in[pi++];
        }
    };
    Layer sa1L[2];  { int d[] = {6, 128, 128};   take(sa1L, 2, d); }
    Layer fe1L[1];  { int d[] = {384, 128};      take(fe1L, 1, d); }
    Layer sa2L[2];  { int d[] = {256, 256, 256}; take(sa2L, 2, d); }
    Layer fe2L[1];  { int d[] = {768, 256};      take(fe2L, 1, d); }
    Layer sa3L[1];  { int d[] = {512, 512};      take(sa3L, 1, d); }
    Layer su1aL[1]; { int d[] = {768, 512};      take(su1aL, 1, d); }
    Layer su1bL[1]; { int d[] = {768, 512};      take(su1bL, 1, d); }
    Layer su2aL[1]; { int d[] = {640, 512};      take(su2aL, 1, d); }
    Layer su2bL[1]; { int d[] = {640, 512};      take(su2bL, 1, d); }
    Layer fpL[1];   { int d[] = {515, 256};      take(fpL, 1, d); }
    const float* hW1   = (const float*)d_in[pi++];
    const float* hb1   = (const float*)d_in[pi++];
    const float* hg    = (const float*)d_in[pi++];
    const float* hbeta = (const float*)d_in[pi++];
    const float* hW2   = (const float*)d_in[pi++];
    const float* hb2   = (const float*)d_in[pi++];

    // --- workspace bump allocator (assumes ws_size large enough, ~580MB)
    char* base = (char*)d_ws; size_t off = 0;
    auto allocf = [&](size_t n) -> float* {
        float* p = (float*)(base + off); off += ((n * 4 + 255) & ~(size_t)255); return p;
    };
    auto alloch = [&](size_t n) -> _Float16* {
        _Float16* p = (_Float16*)(base + off); off += ((n * 2 + 255) & ~(size_t)255); return p;
    };
    auto alloci = [&](size_t n) -> int* {
        int* p = (int*)(base + off); off += ((n * 4 + 255) & ~(size_t)255); return p;
    };

    float*    SA    = allocf(57000000);   // big ping (su2 feat: 640*88000)
    float*    SB    = allocf(46000000);   // big pong (su2 layer out: 512*88000)
    _Float16* XT    = alloch(57000000);   // f16 transposed GEMM operand
    _Float16* W16   = alloch(400000);     // f16 padded weights (max 512*768)
    float*    stats = allocf(1024);
    float*    NXY   = allocf(360000);     // new_xyz temp (max 128*1375)
    int*      FIDX  = alloci(5600);
    int*      GIDX  = alloci(140000);     // max 5500*24
    float* l1a[4]; for (int i = 0; i < 4; ++i) l1a[i] = allocf(128 * 5500);
    float* l1a_pc3 = allocf(3 * 5500);
    float* l1b0 = allocf(128 * 5500);
    float* l1b1 = allocf(128 * 5500);
    float* l2a0 = allocf(256 * 1375);
    float* l2a1 = allocf(256 * 1375);
    float* l2a_pc1 = allocf(128 * 1375);
    float* l2b  = allocf(256 * 1375);
    float* l3pc = allocf(256 * 275);
    float* l3ft = allocf(512 * 275);
    float* l2f  = allocf(512 * 1375);
    float* l1f  = allocf(512 * 5500);
    float* hbuf = allocf(128 * 8192);

    // --- launch helpers
    auto gemm_bias = [&](const float* W, const float* b, const float* X, float* Y,
                         int Co, int Ci, int M) {
        int CiPad = ceilN(Ci, 32), CoPad = ceilN(Co, 64), Mpad = ceilN(M, 64);
        {   int tot = CoPad * CiPad;
            k_cvt_w<<<(tot + 255) / 256, 256, 0, stream>>>(W, W16, Co, Ci, CoPad, CiPad); }
        {   size_t tot = (size_t)Mpad * CiPad;
            k_cvt_xt<<<dim3((unsigned)((tot + 255) / 256)), 256, 0, stream>>>(
                X, XT, Ci, M, CiPad, Mpad); }
        dim3 g(Mpad / 64, CoPad / 64);      // wave: 1 m-tile x 4 co-tiles
        k_wmma_gemm_f16<<<g, dim3(128), 0, stream>>>(W16, XT, b, Y, Co, CiPad, M);
    };
    auto bn = [&](const Layer& L, float* Y, int M, int act) {
        k_bn_stats<<<L.co, 256, 0, stream>>>(Y, M, stats);
        k_bn_apply<<<dim3((M + 255) / 256, L.co), 256, 0, stream>>>(Y, M, stats, L.g, L.beta, act);
    };
    // ping-pong MLP: input must be SA; returns pointer holding final output
    auto mlp = [&](Layer* Ls, int nL, float* X, int M) -> float* {
        float* bufs[2] = {SB, SA};
        float* cur = X;
        for (int l = 0; l < nL; ++l) {
            float* out = bufs[l & 1];
            gemm_bias(Ls[l].W, Ls[l].b, cur, out, Ls[l].co, Ls[l].ci, M);
            bn(Ls[l], out, M, 1);
            cur = out;
        }
        return cur;
    };
    auto ggather = [&](const float* src, int C, int N, int Nq, int K, float* dst,
                       int sub, const float* center) {
        size_t total = (size_t)C * Nq * K;
        k_group_gather<<<dim3((unsigned)((total + 255) / 256)), 256, 0, stream>>>(
            src, C, N, GIDX, Nq, K, dst, sub, center);
    };

    // --- set_abstraction (points == xyz in every call of the reference)
    auto set_abs = [&](const float* xyz, int C, int N, int npoint, int ns,
                       Layer* Ls, int nL, float* out_ft, float* out_xyz) {
        k_fps<<<1, 1024, 0, stream>>>(xyz, C, N, npoint, FIDX);
        k_gather_cols<<<(C * npoint + 255) / 256, 256, 0, stream>>>(xyz, C, N, FIDX, npoint, NXY);
        k_ball<<<npoint, 256, 0, stream>>>(xyz, C, N, NXY, npoint, 100.f * 100.f, ns, GIDX);
        size_t M = (size_t)npoint * ns;
        ggather(xyz, C, N, npoint, ns, SA, 1, NXY);                    // g_xyz
        ggather(xyz, C, N, npoint, ns, SA + (size_t)C * M, 0, nullptr);// grouped points
        float* y = mlp(Ls, nL, SA, (int)M);
        int Co = Ls[nL - 1].co;
        k_maxK<<<(Co * npoint + 255) / 256, 256, 0, stream>>>(y, Co, npoint, ns, out_ft);
        if (out_xyz)
            hipMemcpyAsync(out_xyz, NXY, (size_t)C * npoint * 4,
                           hipMemcpyDeviceToDevice, stream);
    };

    // --- flow_embedding (pos==feat in every call)
    auto flow_emb = [&](const float* F1, const float* F2, int C, int N, int K,
                        Layer* Ls, float* out) {
        k_knn<<<N, 256, 0, stream>>>(F1, F2, C, N, N, K, GIDX);
        size_t M = (size_t)N * K;
        ggather(F2, C, N, N, K, SA, 1, F1);                             // pos_diff
        ggather(F2, C, N, N, K, SA + (size_t)C * M, 0, nullptr);        // f2 grouped
        size_t total = (size_t)C * M;
        k_broadcastK<<<dim3((unsigned)((total + 255) / 256)), 256, 0, stream>>>(
            F1, C, N, K, SA + 2 * (size_t)C * M);                       // f1 expanded
        float* y = mlp(Ls, 1, SA, (int)M);
        k_maxK<<<(Ls[0].co * N + 255) / 256, 256, 0, stream>>>(y, Ls[0].co, N, K, out);
    };

    // --- set_upconv
    auto set_upconv = [&](const float* pos1, const float* pos2, const float* feat1,
                          int Cf1, const float* feat2, int Cf2, int C,
                          int N1, int N2, int K, Layer* La, Layer* Lb, float* out) {
        k_knn<<<N1, 256, 0, stream>>>(pos1, pos2, C, N1, N2, K, GIDX);
        size_t M = (size_t)N1 * K;
        ggather(feat2, Cf2, N2, N1, K, SA, 0, nullptr);                 // grouped feat2
        ggather(pos2, C, N2, N1, K, SA + (size_t)Cf2 * M, 1, pos1);     // pos_diff
        float* y = mlp(La, 1, SA, (int)M);                              // -> SB
        int Co = La[0].co;
        k_maxK<<<(Co * N1 + 255) / 256, 256, 0, stream>>>(y, Co, N1, K, SA);
        hipMemcpyAsync(SA + (size_t)Co * N1, feat1, (size_t)Cf1 * N1 * 4,
                       hipMemcpyDeviceToDevice, stream);                // concat feat1
        float* y2 = mlp(Lb, 1, SA, N1);                                 // -> SB
        hipMemcpyAsync(out, y2, (size_t)Lb[0].co * N1 * 4,
                       hipMemcpyDeviceToDevice, stream);
    };

    // ======================= forward pass =======================
    set_abs(x0, 3, 8192, 5500, 16, sa1L, 2, l1a[0], nullptr);
    set_abs(x1, 3, 8192, 5500, 16, sa1L, 2, l1a[1], nullptr);
    set_abs(x2, 3, 8192, 5500, 16, sa1L, 2, l1a[2], nullptr);
    set_abs(x3, 3, 8192, 5500, 16, sa1L, 2, l1a[3], l1a_pc3);

    flow_emb(l1a[1], l1a[0], 128, 5500, 24, fe1L, l1b0);
    flow_emb(l1a[3], l1a[2], 128, 5500, 24, fe1L, l1b1);

    set_abs(l1b0, 128, 5500, 1375, 16, sa2L, 2, l2a0, nullptr);
    set_abs(l1b1, 128, 5500, 1375, 16, sa2L, 2, l2a1, l2a_pc1);

    flow_emb(l2a1, l2a0, 256, 1375, 24, fe2L, l2b);

    set_abs(l2b, 256, 1375, 275, 16, sa3L, 1, l3ft, l3pc);

    set_upconv(l2b, l3pc, l2b, 256, l3ft, 512, 256, 1375, 275, 16, su1aL, su1bL, l2f);
    set_upconv(l1b1, l2a_pc1, l1b1, 128, l2f, 512, 128, 5500, 1375, 16, su2aL, su2bL, l1f);

    // feature propagation: 3-NN interp of l1f onto the 8192 input points + x3
    k_interp3<<<8192, 256, 0, stream>>>(x3, l1a_pc3, l1f, 8192, 5500, 512, SA);
    hipMemcpyAsync(SA + (size_t)512 * 8192, x3, (size_t)3 * 8192 * 4,
                   hipMemcpyDeviceToDevice, stream);
    float* l0 = mlp(fpL, 1, SA, 8192);          // -> SB, (256 x 8192)

    // head: conv -> leaky relu -> BN -> conv -> residual add
    gemm_bias(hW1, hb1, l0, hbuf, 128, 256, 8192);
    k_leaky<<<(128 * 8192 + 255) / 256, 256, 0, stream>>>(hbuf, 128 * 8192);
    k_bn_stats<<<128, 256, 0, stream>>>(hbuf, 8192, stats);
    k_bn_apply<<<dim3(32, 128), 256, 0, stream>>>(hbuf, 8192, stats, hg, hbeta, 0);
    gemm_bias(hW2, hb2, hbuf, SA, 3, 128, 8192);
    k_add<<<(3 * 8192 + 255) / 256, 256, 0, stream>>>(x3, SA, (float*)d_out, 3 * 8192);
}
